// StyleLoss_63522566307939
// MI455X (gfx1250) — matrix operand order repaired and checked by reading, not compiled, not run on hardware
//
#include <hip/hip_runtime.h>
#include <hip/hip_bf16.h>

// ---------------------------------------------------------------------------
// StyleLoss on MI455X (gfx1250, wave32, WMMA).
//
// Dominant work: sim = style_patches^T (7225 x 2331) * img_patches (2331 x 7225)
//   = 2.43e11 FLOP, fused with per-row scaled argmax (sim never materialized).
// Both bf16 hi/lo operand sets (~135 MB) fit in the 192 MB L2 -> compute bound
// -> v_wmma_f32_16x16x32_bf16 with bf16x3 error compensation (3 WMMA per K=32
// block). Inner loop is term-major so consecutive WMMAs hit different
// accumulators (independent C->D chains, no register-rotation copies).
// Loss is recomputed with exact fp32 gathers from the original tensors.
// ---------------------------------------------------------------------------

#define NPATCH   7225         // 85*85 patch positions
#define NPAD     7232         // padded to 452*16
#define NTILES   452
#define KDIM     2331         // 259 channels * 9
#define KPAD     2336         // padded to 73*32
#define KBLOCKS  73
#define NW       85

typedef __bf16 v8bf  __attribute__((ext_vector_type(8)));
typedef __bf16 v16bf __attribute__((ext_vector_type(16)));
typedef float  v8f   __attribute__((ext_vector_type(8)));

union BF16x16 { v16bf v; v8bf h[2]; };

// ---------------------------------------------------------------------------
// 4x4 average pool: (3,1024,1024) -> (3,256,256)
// ---------------------------------------------------------------------------
__global__ void pool4_kernel(const float* __restrict__ src, float* __restrict__ dst) {
    int idx = blockIdx.x * 256 + threadIdx.x;
    if (idx >= 3 * 256 * 256) return;
    int c = idx >> 16;
    int y = (idx >> 8) & 255;
    int x = idx & 255;
    const float* p = src + ((size_t)c << 20) + ((size_t)(y * 4) << 10) + (size_t)(x * 4);
    float s = 0.f;
    #pragma unroll
    for (int a = 0; a < 4; ++a)
        #pragma unroll
        for (int b = 0; b < 4; ++b)
            s += p[a * 1024 + b];
    dst[idx] = s * 0.0625f;
}

// ---------------------------------------------------------------------------
// Build padded, K-contiguous bf16 hi/lo patch matrix (NPAD rows x KPAD cols,
// row = patch index, col = k) and per-patch 1/norm. One block per patch.
// ---------------------------------------------------------------------------
__global__ void build_patches_kernel(const float* __restrict__ resp,   // 256x256x256
                                     const float* __restrict__ pool,   // 3x256x256 (unscaled)
                                     __bf16* __restrict__ hiP,
                                     __bf16* __restrict__ loP,
                                     float* __restrict__ norminv) {
    int i = blockIdx.x;        // 0..NPAD-1
    int t = threadIdx.x;       // 256 threads
    float ssum = 0.f;
    size_t rowbase = (size_t)i * KPAD;
    if (i < NPATCH) {
        int py = i / NW, px = i - py * NW;
        int yb = py * 3, xb = px * 3;
        for (int d = t; d < KPAD; d += 256) {
            float v = 0.f;
            if (d < KDIM) {
                int c   = d / 9;
                int rem = d - c * 9;
                int dy  = rem / 3, dx = rem - dy * 3;
                int y = yb + dy, x = xb + dx;
                if (c < 256) v = resp[((size_t)c << 16) + (y << 8) + x];
                else         v = 50.0f * pool[((size_t)(c - 256) << 16) + (y << 8) + x];
            }
            __bf16 h = (__bf16)v;
            __bf16 l = (__bf16)(v - (float)h);
            hiP[rowbase + d] = h;
            loP[rowbase + d] = l;
            ssum += v * v;
        }
    } else {
        for (int d = t; d < KPAD; d += 256) {
            hiP[rowbase + d] = (__bf16)0.f;
            loP[rowbase + d] = (__bf16)0.f;
        }
    }
    __shared__ float red[256];
    red[t] = ssum;
    __syncthreads();
    for (int s = 128; s > 0; s >>= 1) {
        if (t < s) red[t] += red[t + s];
        __syncthreads();
    }
    if (t == 0) norminv[i] = (i < NPATCH) ? rsqrtf(red[0]) : 0.f;
}

// ---------------------------------------------------------------------------
// Fused GEMM + scaled argmax.
//   C[i,j] = sum_k style[k,i]*img[k,j];  nearest[i] = argmax_j C[i,j]*sninv[j]
// One block (8 wave32) per 16-row tile; each wave owns disjoint groups of
// 4 column tiles with a private running (max,argmax); LDS reduction at end.
// ---------------------------------------------------------------------------
__global__ __launch_bounds__(256) void gemm_argmax_kernel(
    const __bf16* __restrict__ Ahi, const __bf16* __restrict__ Alo,
    const __bf16* __restrict__ Bhi, const __bf16* __restrict__ Blo,
    const float* __restrict__ sninv, int* __restrict__ nearest) {
    int ti   = blockIdx.x;           // 0..NTILES-1
    int t    = threadIdx.x;
    int w    = t >> 5;
    int lane = t & 31;
    int half = lane >> 4;
    int mn   = lane & 15;            // M row (A) and N col (B/C) within tile
    int i0   = ti * 16;

    // A fragment per ISA layout: lanes 0-15 hold K {0..7,16..23}, lanes 16-31
    // hold K {8..15,24..31} of each 32-wide K block, for row M = lane&15.
    const __bf16* aHiRow = Ahi + (size_t)(i0 + mn) * KPAD + half * 8;
    const __bf16* aLoRow = Alo + (size_t)(i0 + mn) * KPAD + half * 8;

    float best[8];
    int   bidx[8];
    #pragma unroll
    for (int v = 0; v < 8; ++v) { best[v] = -3.4e38f; bidx[v] = 0; }

    v8f zero;
    #pragma unroll
    for (int q = 0; q < 8; ++q) zero[q] = 0.f;

    for (int g = w; g < 113; g += 8) {       // 113 groups of 4 column tiles
        int j0 = g * 64;
        // B fragment base: lanes 0-15 hold K {0..7,16..23} (N=lane), lanes 16-31 K {8..15,24..31}.
        const __bf16* bHiBase = Bhi + (size_t)(j0 + mn) * KPAD + half * 8;
        const __bf16* bLoBase = Blo + (size_t)(j0 + mn) * KPAD + half * 8;

        v8f acc[4];
        #pragma unroll
        for (int jt = 0; jt < 4; ++jt) acc[jt] = zero;

        #pragma unroll 1
        for (int kb = 0; kb < KBLOCKS; ++kb) {
            size_t ko = (size_t)kb * 32;
            __builtin_prefetch(aHiRow + ko + 64, 0, 1);

            BF16x16 ah, al;
            ah.h[0] = *(const v8bf*)(aHiRow + ko);
            ah.h[1] = *(const v8bf*)(aHiRow + ko + 16);
            al.h[0] = *(const v8bf*)(aLoRow + ko);
            al.h[1] = *(const v8bf*)(aLoRow + ko + 16);

            BF16x16 bh[4], bl[4];
            #pragma unroll
            for (int jt = 0; jt < 4; ++jt) {
                const __bf16* ph = bHiBase + (size_t)jt * 16 * KPAD + ko;
                const __bf16* pl = bLoBase + (size_t)jt * 16 * KPAD + ko;
                bh[jt].h[0] = *(const v8bf*)(ph);
                bh[jt].h[1] = *(const v8bf*)(ph + 16);
                bl[jt].h[0] = *(const v8bf*)(pl);
                bl[jt].h[1] = *(const v8bf*)(pl + 16);
            }

            // Term-major issue order: consecutive WMMAs use different
            // accumulators -> 4 independent C->D chains fill the matrix pipe.
            #pragma unroll
            for (int jt = 0; jt < 4; ++jt)
                acc[jt] = __builtin_amdgcn_wmma_f32_16x16x32_bf16(
                    false, ah.v, false, bh[jt].v, (short)0, acc[jt], false, false);
            #pragma unroll
            for (int jt = 0; jt < 4; ++jt)
                acc[jt] = __builtin_amdgcn_wmma_f32_16x16x32_bf16(
                    false, ah.v, false, bl[jt].v, (short)0, acc[jt], false, false);
            #pragma unroll
            for (int jt = 0; jt < 4; ++jt)
                acc[jt] = __builtin_amdgcn_wmma_f32_16x16x32_bf16(
                    false, al.v, false, bh[jt].v, (short)0, acc[jt], false, false);
        }
        // Epilogue: C/D layout -> lane holds rows M = v + 8*half, col N = lane&15.
        #pragma unroll
        for (int jt = 0; jt < 4; ++jt) {
            int j = j0 + jt * 16 + mn;
            if (j < NPATCH) {
                float sc = sninv[j];
                #pragma unroll
                for (int v = 0; v < 8; ++v) {
                    float cand = acc[jt][v] * sc;
                    if (cand > best[v]) { best[v] = cand; bidx[v] = j; }
                }
            }
        }
    }

    __shared__ float sval[8 * 32 * 8];
    __shared__ int   sidx[8 * 32 * 8];
    #pragma unroll
    for (int v = 0; v < 8; ++v) {
        sval[(w * 32 + lane) * 8 + v] = best[v];
        sidx[(w * 32 + lane) * 8 + v] = bidx[v];
    }
    __syncthreads();
    if (t < 16) {
        int r = t;                               // row within tile
        float bb = -3.4e38f; int bj = 0;
        for (int ww = 0; ww < 8; ++ww)
            for (int ln = 0; ln < 16; ++ln) {
                int L = (r < 8) ? ln : (16 + ln);
                int e = (ww * 32 + L) * 8 + (r & 7);
                float val = sval[e];
                if (val > bb) { bb = val; bj = sidx[e]; }
            }
        if (i0 + r < NPATCH) nearest[i0 + r] = bj;
    }
}

// ---------------------------------------------------------------------------
// Exact fp32 loss: per patch i, sum_d (img[d,i] - style[d,nearest[i]])^2,
// gathered straight from the original tensors. One block per patch.
// ---------------------------------------------------------------------------
__global__ void loss_partial_kernel(const float* __restrict__ mresp,
                                    const float* __restrict__ opool,
                                    const float* __restrict__ sresp,
                                    const float* __restrict__ spool,
                                    const int* __restrict__ nearest,
                                    float* __restrict__ partial) {
    int i = blockIdx.x;      // 0..NPATCH-1
    int t = threadIdx.x;
    __shared__ int sj;
    if (t == 0) sj = nearest[i];
    __syncthreads();
    int j  = sj;
    int py = i / NW, px = i - py * NW;
    int qy = j / NW, qx = j - qy * NW;
    int iy0 = py * 3, ix0 = px * 3, jy0 = qy * 3, jx0 = qx * 3;
    float s = 0.f;
    for (int d = t; d < KDIM; d += 256) {
        int c   = d / 9;
        int rem = d - c * 9;
        int dy  = rem / 3, dx = rem - dy * 3;
        float a, b;
        if (c < 256) {
            a = mresp[((size_t)c << 16) + ((iy0 + dy) << 8) + (ix0 + dx)];
            b = sresp[((size_t)c << 16) + ((jy0 + dy) << 8) + (jx0 + dx)];
        } else {
            a = 50.f * opool[((size_t)(c - 256) << 16) + ((iy0 + dy) << 8) + (ix0 + dx)];
            b = 50.f * spool[((size_t)(c - 256) << 16) + ((jy0 + dy) << 8) + (jx0 + dx)];
        }
        float df = a - b;
        s += df * df;
    }
    __shared__ float red[256];
    red[t] = s;
    __syncthreads();
    for (int st = 128; st > 0; st >>= 1) {
        if (t < st) red[t] += red[t + st];
        __syncthreads();
    }
    if (t == 0) partial[i] = red[0];
}

// Deterministic fixed-order final reduction (no float atomics).
__global__ void loss_final_kernel(const float* __restrict__ partial,
                                  float* __restrict__ out) {
    int t = threadIdx.x;
    float s = 0.f;
    for (int k = t; k < NPATCH; k += 256) s += partial[k];
    __shared__ float red[256];
    red[t] = s;
    __syncthreads();
    for (int st = 128; st > 0; st >>= 1) {
        if (t < st) red[t] += red[t + st];
        __syncthreads();
    }
    if (t == 0) out[0] = red[0] / (float)((double)KDIM * (double)NPATCH);
}

// ---------------------------------------------------------------------------
extern "C" void kernel_launch(void* const* d_in, const int* in_sizes, int n_in,
                              void* d_out, int out_size, void* d_ws, size_t ws_size,
                              hipStream_t stream) {
    const float* style_resp = (const float*)d_in[0];   // (1,256,256,256)
    const float* style_map  = (const float*)d_in[1];   // (1,3,1024,1024)
    const float* output_map = (const float*)d_in[2];   // (1,3,1024,1024)
    const float* model_resp = (const float*)d_in[3];   // (1,256,256,256)
    float* out = (float*)d_out;

    char*  ws  = (char*)d_ws;
    size_t off = 0;
    auto carve = [&](size_t bytes) -> void* {
        void* p = ws + off;
        off = (off + bytes + 255) & ~(size_t)255;
        return p;
    };

    const size_t planeB = (size_t)NPAD * KPAD * sizeof(__bf16);   // ~33.8 MB
    __bf16* ST_hi = (__bf16*)carve(planeB);
    __bf16* ST_lo = (__bf16*)carve(planeB);
    __bf16* IT_hi = (__bf16*)carve(planeB);
    __bf16* IT_lo = (__bf16*)carve(planeB);
    float*  spool   = (float*)carve((size_t)3 * 256 * 256 * sizeof(float));
    float*  opool   = (float*)carve((size_t)3 * 256 * 256 * sizeof(float));
    float*  sninv   = (float*)carve((size_t)NPAD * sizeof(float));
    float*  dummyn  = (float*)carve((size_t)NPAD * sizeof(float));
    int*    nearest = (int*)  carve((size_t)NPAD * sizeof(int));
    float*  partial = (float*)carve((size_t)NPAD * sizeof(float));

    int poolN = 3 * 256 * 256;
    pool4_kernel<<<(poolN + 255) / 256, 256, 0, stream>>>(style_map, spool);
    pool4_kernel<<<(poolN + 255) / 256, 256, 0, stream>>>(output_map, opool);

    build_patches_kernel<<<NPAD, 256, 0, stream>>>(style_resp, spool, ST_hi, ST_lo, sninv);
    build_patches_kernel<<<NPAD, 256, 0, stream>>>(model_resp, opool, IT_hi, IT_lo, dummyn);

    gemm_argmax_kernel<<<NTILES, 256, 0, stream>>>(ST_hi, ST_lo, IT_hi, IT_lo, sninv, nearest);

    loss_partial_kernel<<<NPATCH, 256, 0, stream>>>(model_resp, opool, style_resp, spool,
                                                    nearest, partial);
    loss_final_kernel<<<1, 256, 0, stream>>>(partial, out);
}
